// TTRNN_5179730559369
// MI455X (gfx1250) — compile-verified, matrix operand order checked by reading, not run on hardware
//
#include <hip/hip_runtime.h>
#include <stdint.h>

typedef __attribute__((ext_vector_type(16))) __bf16 v16bf;
typedef __attribute__((ext_vector_type(8)))  float  v8f;
typedef unsigned short u16;
typedef unsigned int   u32;

union V16 { v16bf v; struct { uint4 lo, hi; } q; };

__device__ __forceinline__ u16 f2bf(float f) {
    u32 u = __float_as_uint(f);
    u32 r = (u + 0x7FFFu + ((u >> 16) & 1u)) >> 16;   // round-to-nearest-even
    return (u16)r;
}
__device__ __forceinline__ float bf2f(u16 h) { return __uint_as_float(((u32)h) << 16); }
__device__ __forceinline__ float sigf(float x) { return 1.0f / (1.0f + __expf(-x)); }

struct CorePtrs { const float* c0[16]; const float* c1[16]; const float* c2[16]; const float* c3[16]; };
struct BiasPtrs { const float* p[8]; };

// ---- TT dense materialization: T2[m12,n12,r2] = sum_r1 C1[m1,n1,r1]*C2[r1,m2,n2,r2]
__global__ void k_tt_t2(CorePtrs cp, float* __restrict__ T2) {
    const int w = blockIdx.x;
    const float* c0 = cp.c0[w];
    const float* c1 = cp.c1[w];
    for (int idx = threadIdx.x; idx < 16384; idx += 256) {
        const int r2 = idx & 15, n12 = (idx >> 4) & 31, m12 = idx >> 9;
        const int m1 = m12 >> 3, m2 = m12 & 7, n1 = n12 >> 3, n2 = n12 & 7;
        float s = 0.f;
        #pragma unroll
        for (int r1 = 0; r1 < 16; ++r1)
            s = fmaf(c0[(m1 * 4 + n1) * 16 + r1], c1[(((r1 * 8 + m2) * 8 + n2) * 16) + r2], s);
        T2[(size_t)w * 16384 + idx] = s;
    }
}

// ---- T3[m123,n123,r3] = sum_r2 T2[m12,n12,r2]*C3[r2,m3,n3,r3]
__global__ void k_tt_t3(CorePtrs cp, const float* __restrict__ T2, float* __restrict__ T3) {
    const int gid = blockIdx.x * 256 + threadIdx.x;      // 16 * 65536
    const int w = gid >> 16, rem = gid & 65535;
    const int m123 = rem >> 8, n123 = rem & 255;
    const int m12 = m123 >> 3, m3 = m123 & 7, n12 = n123 >> 3, n3 = n123 & 7;
    const float* c2 = cp.c2[w];
    const float* t2 = T2 + (size_t)w * 16384 + (m12 * 32 + n12) * 16;
    float t2l[16];
    #pragma unroll
    for (int r2 = 0; r2 < 16; ++r2) t2l[r2] = t2[r2];
    float* out = T3 + (size_t)gid * 16;
    #pragma unroll
    for (int r3 = 0; r3 < 16; ++r3) {
        float s = 0.f;
        #pragma unroll
        for (int r2 = 0; r2 < 16; ++r2)
            s = fmaf(t2l[r2], c2[((r2 * 8 + m3) * 8 + n3) * 16 + r3], s);
        out[r3] = s;
    }
}

// ---- Dense Wt[out_n][in_m] bf16: W = sum_r3 T3[m123,n123,r3]*C4[r3,m4,n4]
__global__ void k_tt_w(CorePtrs cp, const float* __restrict__ T3, u16* __restrict__ Wd) {
    const u32 gid = blockIdx.x * 256 + threadIdx.x;      // 16 * 1048576
    const int w = gid >> 20;
    const u32 rem = gid & 0xFFFFFu;
    const int n = rem >> 10, m = rem & 1023;
    const int m123 = m >> 2, m4 = m & 3, n123 = n >> 2, n4 = n & 3;
    const float* c3 = cp.c3[w];
    const float* t3 = T3 + ((size_t)w * 65536 + (size_t)m123 * 256 + n123) * 16;
    float s = 0.f;
    #pragma unroll
    for (int r3 = 0; r3 < 16; ++r3)
        s = fmaf(t3[r3], c3[(r3 * 4 + m4) * 4 + n4], s);
    const int l = w >> 3, t = (w >> 2) & 1, g = w & 3;
    // layout: [(l*2+t)][gate*1024 + n][m], row-major over m (K)
    Wd[(((size_t)(l * 2 + t) * 4096) + (size_t)g * 1024 + n) * 1024 + m] = f2bf(s);
}

// ---- bias pack: bias[l*4096 + g*1024 + j], gate order i,f,g,o
__global__ void k_bias(BiasPtrs bp, float* __restrict__ out) {
    const int gid = blockIdx.x * 256 + threadIdx.x;      // 8192
    const int l = gid >> 12, r = gid & 4095;
    const int g = r >> 10, j = r & 1023;
    out[gid] = bp.p[l * 4 + g][j];
}

// ---- prenet: res[t*64+b][k] = leakyrelu(x[b,t,:] @ W + b), bf16, time-major
__global__ void k_prenet(const float* __restrict__ x, const float* __restrict__ pw,
                         const float* __restrict__ pb, u16* __restrict__ res) {
    __shared__ float xs[304];
    const int bid = blockIdx.x;                 // = b*256 + t
    const int b = bid >> 8, t = bid & 255;
    const int tid = threadIdx.x;
    for (int f = tid; f < 300; f += 256) xs[f] = x[(size_t)bid * 300 + f];
    __syncthreads();
    #pragma unroll
    for (int j = 0; j < 4; ++j) {
        const int k = tid + j * 256;
        float s = pb[k];
        for (int f = 0; f < 300; ++f) s = fmaf(xs[f], pw[(size_t)f * 1024 + k], s);
        s = s >= 0.f ? s : 0.1f * s;
        res[((size_t)(t * 64 + b)) * 1024 + k] = f2bf(s);
    }
}

__global__ void k_zero(u32* __restrict__ p, int n) {
    int i = blockIdx.x * 256 + threadIdx.x;
    if (i < n) p[i] = 0;
}

// ======== LSTM step GEMM machinery ========
struct BTiles { V16 a0, a1, b0, b1, b2, b3; };

__device__ __forceinline__ void load_tiles(BTiles& T,
    const u16* xr0, const u16* xr1,
    const u16* w0, const u16* w1, const u16* w2, const u16* w3, int c0) {
    const int c1 = c0 + 16;
    T.a0.q.lo = *(const uint4*)(xr0 + c0); T.a0.q.hi = *(const uint4*)(xr0 + c1);
    T.a1.q.lo = *(const uint4*)(xr1 + c0); T.a1.q.hi = *(const uint4*)(xr1 + c1);
    T.b0.q.lo = *(const uint4*)(w0 + c0);  T.b0.q.hi = *(const uint4*)(w0 + c1);
    T.b1.q.lo = *(const uint4*)(w1 + c0);  T.b1.q.hi = *(const uint4*)(w1 + c1);
    T.b2.q.lo = *(const uint4*)(w2 + c0);  T.b2.q.hi = *(const uint4*)(w2 + c1);
    T.b3.q.lo = *(const uint4*)(w3 + c0);  T.b3.q.hi = *(const uint4*)(w3 + c1);
}

#define WMMA_BF16(A, B, Cacc, RA, RB) \
    __builtin_amdgcn_wmma_f32_16x16x32_bf16(false, (A), false, (B), (short)0, (Cacc), (RA), (RB))

// Serpentine order: adjacent WMMAs share one operand; set matching reuse hint.
__device__ __forceinline__ void mma8(const BTiles& T, v8f acc[2][4]) {
    acc[0][0] = WMMA_BF16(T.a0.v, T.b0.v, acc[0][0], false, true );
    acc[1][0] = WMMA_BF16(T.a1.v, T.b0.v, acc[1][0], true,  false);
    acc[1][1] = WMMA_BF16(T.a1.v, T.b1.v, acc[1][1], false, true );
    acc[0][1] = WMMA_BF16(T.a0.v, T.b1.v, acc[0][1], true,  false);
    acc[0][2] = WMMA_BF16(T.a0.v, T.b2.v, acc[0][2], false, true );
    acc[1][2] = WMMA_BF16(T.a1.v, T.b2.v, acc[1][2], true,  false);
    acc[1][3] = WMMA_BF16(T.a1.v, T.b3.v, acc[1][3], false, true );
    acc[0][3] = WMMA_BF16(T.a0.v, T.b3.v, acc[0][3], false, false);
}

// K=1024 GEMM phase, 2-deep software pipeline so next-tile loads overlap WMMAs.
__device__ __forceinline__ void gemm_phase(
    const u16* __restrict__ xr0, const u16* __restrict__ xr1,
    const u16* __restrict__ w0, const u16* __restrict__ w1,
    const u16* __restrict__ w2, const u16* __restrict__ w3,
    int kg, v8f acc[2][4]) {
    BTiles cur, nxt;
    load_tiles(cur, xr0, xr1, w0, w1, w2, w3, kg * 8);
    #pragma unroll 2
    for (int kk = 0; kk < 31; ++kk) {
        const int c0 = (kk + 1) * 32 + kg * 8;
        load_tiles(nxt, xr0, xr1, w0, w1, w2, w3, c0);
        __builtin_prefetch((const void*)(xr0 + c0 + 256), 0, 1);
        __builtin_prefetch((const void*)(xr1 + c0 + 256), 0, 1);
        mma8(cur, acc);
        cur = nxt;
    }
    mma8(cur, acc);
}

// ---- one LSTM layer, one timestep: G = X@Wih^T + H@Whh^T + b; elementwise update.
// 128 waves; each wave owns TWO 16x16 m-tiles for one j-tile and all 4 gates,
// so every B tile feeds 2 WMMAs (12 b128 loads : 8 WMMAs).
__global__ void __launch_bounds__(128) k_lstm_step(
    const u16* __restrict__ X, const u16* __restrict__ H,
    const u16* __restrict__ Wih, const u16* __restrict__ Whh,
    const float* __restrict__ bias, float* __restrict__ C, u16* __restrict__ Hout)
{
    const int tid  = threadIdx.x;
    const int wave = blockIdx.x * 4 + (tid >> 5);   // 0..127
    const int lane = tid & 31;
    const int ln   = lane & 15;     // row within tile (A: M, B: N)
    const int kg   = lane >> 4;     // K-group select per documented 16-bit layout
    const int mBase = (wave >> 6) * 32;   // 0 or 32
    const int jBase = (wave & 63) * 16;   // 0..1008

    v8f acc[2][4] = {};

    gemm_phase(X + (size_t)(mBase + ln) * 1024,
               X + (size_t)(mBase + 16 + ln) * 1024,
               Wih + ((size_t)(0 * 1024 + jBase + ln)) * 1024,
               Wih + ((size_t)(1 * 1024 + jBase + ln)) * 1024,
               Wih + ((size_t)(2 * 1024 + jBase + ln)) * 1024,
               Wih + ((size_t)(3 * 1024 + jBase + ln)) * 1024, kg, acc);
    gemm_phase(H + (size_t)(mBase + ln) * 1024,
               H + (size_t)(mBase + 16 + ln) * 1024,
               Whh + ((size_t)(0 * 1024 + jBase + ln)) * 1024,
               Whh + ((size_t)(1 * 1024 + jBase + ln)) * 1024,
               Whh + ((size_t)(2 * 1024 + jBase + ln)) * 1024,
               Whh + ((size_t)(3 * 1024 + jBase + ln)) * 1024, kg, acc);

    // D layout: lane -> N = jBase+ln ; VGPR v -> M = mBase + half*16 + kg*8 + v
    const int n = jBase + ln;
    const float bi  = bias[n], bff = bias[1024 + n], bgg = bias[2048 + n], boo = bias[3072 + n];
    #pragma unroll
    for (int half = 0; half < 2; ++half) {
        const int mh = mBase + half * 16 + kg * 8;
        #pragma unroll
        for (int v = 0; v < 8; ++v) {
            const size_t off = (size_t)(mh + v) * 1024 + n;
            const float gi = sigf (acc[half][0][v] + bi);
            const float gf = sigf (acc[half][1][v] + bff);
            const float gg = tanhf(acc[half][2][v] + bgg);
            const float go = sigf (acc[half][3][v] + boo);
            const float cn = gf * C[off] + gi * gg;
            C[off] = cn;
            Hout[off] = f2bf(go * tanhf(cn));
        }
    }
}

// ---- postnet: out[b,t,c] = ys[t*64+b,:] @ pw[:,c] + pb[c]  (b128 bf16 reads)
__global__ void k_postnet(const u16* __restrict__ ys, const float* __restrict__ pw,
                          const float* __restrict__ pb, float* __restrict__ out) {
    const int gid = blockIdx.x * 256 + threadIdx.x;   // 65536
    const int row = gid >> 2, c = gid & 3;
    const int b = row >> 8, t = row & 255;
    const u16* yr = ys + (size_t)(t * 64 + b) * 1024;
    const float* pwc = pw + c;
    float s = pb[c];
    for (int j = 0; j < 1024; j += 8) {
        uint4 q = *(const uint4*)(yr + j);
        const u16* e = (const u16*)&q;
        #pragma unroll
        for (int u = 0; u < 8; ++u) s = fmaf(bf2f(e[u]), pwc[(j + u) * 4], s);
    }
    out[gid] = s;
}

extern "C" void kernel_launch(void* const* d_in, const int* in_sizes, int n_in,
                              void* d_out, int out_size, void* d_ws, size_t ws_size,
                              hipStream_t stream) {
    (void)in_sizes; (void)n_in; (void)out_size; (void)ws_size;
    // Insertion-order flatten of setup_inputs():
    // 0:x 1:prenet_w 2:prenet_b 3:postnet_w 4:postnet_b
    // then per layer: ih{i,f,g,o}x4cores(16), hh{...}(16), b{i,f,g,o}(4)
    const float* x      = (const float*)d_in[0];
    const float* pre_w  = (const float*)d_in[1];
    const float* pre_b  = (const float*)d_in[2];
    const float* post_w = (const float*)d_in[3];
    const float* post_b = (const float*)d_in[4];

    CorePtrs cp; BiasPtrs bp;
    for (int l = 0; l < 2; ++l) {
        for (int t = 0; t < 2; ++t)
            for (int g = 0; g < 4; ++g) {
                const int w = l * 8 + t * 4 + g;
                const int base = 5 + l * 36 + t * 16 + g * 4;
                cp.c0[w] = (const float*)d_in[base + 0];
                cp.c1[w] = (const float*)d_in[base + 1];
                cp.c2[w] = (const float*)d_in[base + 2];
                cp.c3[w] = (const float*)d_in[base + 3];
            }
        for (int g = 0; g < 4; ++g)
            bp.p[l * 4 + g] = (const float*)d_in[5 + l * 36 + 32 + g];
    }

    uint8_t* ws = (uint8_t*)d_ws;
    u16*   Wd   = (u16*)  (ws + 0ull);            // 32 MB dense bf16 weights (L2-resident)
    float* T2   = (float*)(ws + 33554432ull);     // 1 MB
    float* T3   = (float*)(ws + 34603008ull);     // 64 MB (consumed before res written)
    u16*   res  = (u16*)  (ws + 34603008ull);     // 32 MB, aliases T3 (stream-ordered)
    u16*   ys   = (u16*)  (ws + 101711872ull);    // 32 MB (time-major h1 / postnet input)
    u16*   zh1  = (u16*)  (ws + 135266304ull);    // 128 KB zero h1(-1)
    u16*   h0a  = (u16*)  (ws + 135397376ull);    // 128 KB h0 ping
    u16*   h0b  = (u16*)  (ws + 135528448ull);    // 128 KB h0 pong
    float* C0   = (float*)(ws + 135659520ull);    // 256 KB
    float* C1   = (float*)(ws + 135921664ull);    // 256 KB
    float* bias = (float*)(ws + 136183808ull);    // 32 KB

    k_bias   <<<32,    256, 0, stream>>>(bp, bias);
    k_tt_t2  <<<16,    256, 0, stream>>>(cp, T2);
    k_tt_t3  <<<4096,  256, 0, stream>>>(cp, T2, T3);
    k_tt_w   <<<65536, 256, 0, stream>>>(cp, T3, Wd);
    k_prenet <<<16384, 256, 0, stream>>>(x, pre_w, pre_b, res);
    k_zero   <<<896,   256, 0, stream>>>((u32*)zh1, 229376);   // zh1,h0a,h0b,C0,C1

    const u16* Wih0 = Wd;
    const u16* Whh0 = Wd + 4194304ull;
    const u16* Wih1 = Wd + 8388608ull;
    const u16* Whh1 = Wd + 12582912ull;

    for (int t = 0; t < 256; ++t) {
        const u16* h0p = (t & 1) ? h0a : h0b;
        u16*       h0n = (t & 1) ? h0b : h0a;
        k_lstm_step<<<32, 128, 0, stream>>>(res + (size_t)t * 65536, h0p,
                                            Wih0, Whh0, bias, C0, h0n);
        const u16* h1p = (t == 0) ? zh1 : (ys + (size_t)(t - 1) * 65536);
        k_lstm_step<<<32, 128, 0, stream>>>(h0n, h1p,
                                            Wih1, Whh1, bias + 4096, C1,
                                            ys + (size_t)t * 65536);
    }

    k_postnet<<<256, 256, 0, stream>>>(ys, post_w, post_b, (float*)d_out);
}
